// JRTransformer_31138512896454
// MI455X (gfx1250) — matrix operand order repaired and checked by reading, not compiled, not run on hardware
//
#include <hip/hip_runtime.h>
#include <hip/hip_bf16.h>
#include <math.h>

// ---------------------------------------------------------------------------
// JRTransformer forward for MI455X (gfx1250): all GEMMs via
// v_wmma_f32_16x16x32_f16 (f16 in, f32 accumulate), VALU for LN / softmax /
// gelu / gather kernels. All GEMM K dims are multiples of 32 (ragged inputs
// are zero-padded up front) so fragment loads are branch-free b128 loads.
// ---------------------------------------------------------------------------

typedef __attribute__((ext_vector_type(16))) _Float16 v16h;
typedef __attribute__((ext_vector_type(8)))  _Float16 v8h;
typedef __attribute__((ext_vector_type(8)))  float    v8f;

#define LN_EPS 1e-5f

__device__ __forceinline__ float gelu_exact(float x) {
  return 0.5f * x * (1.0f + erff(x * 0.70710678118654752440f));
}

// A fragment: f32 -> f16, 8 contiguous at p, 8 contiguous at p+16 (ISA layout)
__device__ __forceinline__ v16h load_a_frag(const float* __restrict__ p) {
  const float4 p0 = *(const float4*)(p);
  const float4 p1 = *(const float4*)(p + 4);
  const float4 p2 = *(const float4*)(p + 16);
  const float4 p3 = *(const float4*)(p + 20);
  v16h a;
  a[0]  = (_Float16)p0.x; a[1]  = (_Float16)p0.y; a[2]  = (_Float16)p0.z; a[3]  = (_Float16)p0.w;
  a[4]  = (_Float16)p1.x; a[5]  = (_Float16)p1.y; a[6]  = (_Float16)p1.z; a[7]  = (_Float16)p1.w;
  a[8]  = (_Float16)p2.x; a[9]  = (_Float16)p2.y; a[10] = (_Float16)p2.z; a[11] = (_Float16)p2.w;
  a[12] = (_Float16)p3.x; a[13] = (_Float16)p3.y; a[14] = (_Float16)p3.z; a[15] = (_Float16)p3.w;
  return a;
}

// B fragment: 8 contiguous f16 at p, 8 contiguous at p+16 (ISA layout)
__device__ __forceinline__ v16h load_b_frag(const _Float16* __restrict__ p) {
  v8h b0 = *(const v8h*)(p);
  v8h b1 = *(const v8h*)(p + 16);
  return __builtin_shufflevector(b0, b1, 0, 1, 2, 3, 4, 5, 6, 7,
                                 8, 9, 10, 11, 12, 13, 14, 15);
}

// ------------------------- weight prep -------------------------------------
// f32 W[K,N] -> f16 Wt[Npad,Kpad] (output-channel-major, zero padded)
__global__ void prep_weight(const float* __restrict__ W, _Float16* __restrict__ Wt,
                            int K, int N, int Kpad, int Npad) {
  long i = (long)blockIdx.x * 256 + threadIdx.x;
  long tot = (long)Npad * Kpad;
  if (i >= tot) return;
  int n = (int)(i / Kpad);
  int k = (int)(i % Kpad);
  float v = (n < N && k < K) ? W[(long)k * N + n] : 0.f;
  Wt[i] = (_Float16)v;
}

// pad x_relation rows 18 -> 32 with zeros
__global__ void pad_rows_18_32(const float* __restrict__ x, float* __restrict__ y,
                               long rows) {
  long i = (long)blockIdx.x * 256 + threadIdx.x;
  if (i >= rows * 32) return;
  long r = i >> 5;
  int c = (int)(i & 31);
  y[i] = (c < 18) ? x[r * 18 + c] : 0.f;
}

// ------------------------- WMMA GEMM ---------------------------------------
// C[M,N] = act(A[M,K] * W[K,N] + bias) (+ resid)
// A: f32 row-major, row stride == K (multiple of 32); Wt: f16 [Npad,K].
// One wave -> one 16x16 tile of D; 8 waves per block; pipelined K loop.
__global__ __launch_bounds__(256) void gemm_f16_wmma(
    const float* __restrict__ A, const _Float16* __restrict__ Wt,
    const float* __restrict__ bias, const float* __restrict__ resid,
    float* __restrict__ C, int M, int N, int K, int act) {
  const int lane = threadIdx.x & 31;
  const int wave = threadIdx.x >> 5;
  const long tilesN = (long)((N + 15) >> 4);
  const long tilesM = (long)(M >> 4);
  const long tile = (long)blockIdx.x * 8 + wave;
  if (tile >= tilesM * tilesN) return;
  const int tm = (int)(tile / tilesN);
  const int tn = (int)(tile % tilesN);
  const int l16 = lane & 15;
  const int hi  = lane >> 4;            // lane half select
  const int kgA = hi ? 8 : 0;           // A K sub-offset per ISA layout
  const int kgB = hi << 3;              // B K sub-offset per ISA layout
  const float*    __restrict__ ap = A  + (long)(tm * 16 + l16) * K + kgA;
  const _Float16* __restrict__ bp = Wt + (long)(tn * 16 + l16) * K + kgB;

  v8f acc = {};
  v16h a = load_a_frag(ap);
  v16h b = load_b_frag(bp);
  for (int k0 = 32; k0 < K; k0 += 32) {
    __builtin_prefetch(ap + k0 + 32, 0, 1);     // global_prefetch_b8 (speculative)
    v16h an = load_a_frag(ap + k0);
    v16h bn = load_b_frag(bp + k0);
    acc = __builtin_amdgcn_wmma_f32_16x16x32_f16(false, a, false, b,
                                                 (short)0, acc, false, false);
    a = an; b = bn;
  }
  acc = __builtin_amdgcn_wmma_f32_16x16x32_f16(false, a, false, b,
                                               (short)0, acc, false, false);

  const int ncol = tn * 16 + l16;
  if (ncol < N) {
    const float bv = bias ? bias[ncol] : 0.f;
#pragma unroll
    for (int v = 0; v < 8; ++v) {
      int mr = tm * 16 + v + (hi << 3);         // D: m = vgpr + 8*(lane>=16)
      float x = acc[v] + bv;
      if (act) x = gelu_exact(x);
      long o = (long)mr * N + ncol;
      if (resid) x += resid[o];
      C[o] = x;
    }
  }
}

// ------------------------- LayerNorm (C=128) --------------------------------
__global__ __launch_bounds__(128) void ln_128(const float* __restrict__ x,
                                              const float* __restrict__ g,
                                              const float* __restrict__ b,
                                              float* __restrict__ y) {
  long r = blockIdx.x;
  int t = threadIdx.x;
  float v = x[r * 128 + t];
  __shared__ float sm[128];
  sm[t] = v; __syncthreads();
  for (int s = 64; s > 0; s >>= 1) { if (t < s) sm[t] += sm[t + s]; __syncthreads(); }
  float mean = sm[0] * (1.0f / 128.0f); __syncthreads();
  float d = v - mean;
  sm[t] = d * d; __syncthreads();
  for (int s = 64; s > 0; s >>= 1) { if (t < s) sm[t] += sm[t + s]; __syncthreads(); }
  float var = sm[0] * (1.0f / 128.0f);
  y[r * 128 + t] = d * rsqrtf(var + LN_EPS) * g[t] + b[t];
}

// --------- build rel_in = [rf, rf^T, jf[m], jf[n]] and LN over 4C=512 -------
__global__ __launch_bounds__(256) void build_relln(
    const float* __restrict__ rf, const float* __restrict__ jf,
    const float* __restrict__ g, const float* __restrict__ b,
    float* __restrict__ out, long rowStart) {
  long r = rowStart + blockIdx.x;            // global pair row
  int bi = (int)(r / 5625);
  int rem = (int)(r % 5625);
  int i = rem / 75, j = rem % 75;
  int t = threadIdx.x;

  auto fetch = [&](int c) -> float {
    if (c < 128)      return rf[((long)(bi * 75 + i) * 75 + j) * 128 + c];
    else if (c < 256) return rf[((long)(bi * 75 + j) * 75 + i) * 128 + (c - 128)];
    else if (c < 384) return jf[(long)(bi * 75 + j) * 128 + (c - 256)];  // joint_i -> jf[b,m]
    else              return jf[(long)(bi * 75 + i) * 128 + (c - 384)];  // joint_j -> jf[b,n]
  };
  float v0 = fetch(t), v1 = fetch(t + 256);
  __shared__ float sm[256];
  sm[t] = v0 + v1; __syncthreads();
  for (int s = 128; s > 0; s >>= 1) { if (t < s) sm[t] += sm[t + s]; __syncthreads(); }
  float mean = sm[0] * (1.0f / 512.0f); __syncthreads();
  float d0 = v0 - mean, d1 = v1 - mean;
  sm[t] = d0 * d0 + d1 * d1; __syncthreads();
  for (int s = 128; s > 0; s >>= 1) { if (t < s) sm[t] += sm[t + s]; __syncthreads(); }
  float inv = rsqrtf(sm[0] * (1.0f / 512.0f) + LN_EPS);
  long o = (long)blockIdx.x * 512;
  out[o + t]       = d0 * inv * g[t] + b[t];
  out[o + t + 256] = d1 * inv * g[t + 256] + b[t + 256];
}

// ------------------------- positional encodings -----------------------------
__global__ __launch_bounds__(128) void addpe_j(float* __restrict__ fj,
                                               const float* __restrict__ pep,
                                               const float* __restrict__ pej) {
  int c = threadIdx.x;
  long row = blockIdx.x;                 // b*75+n
  int n = (int)(row % 75);
  fj[row * 128 + c] += pep[(n / 25) * 128 + c] + pej[(n % 25) * 128 + c];
}

__global__ __launch_bounds__(128) void addpe_r(float* __restrict__ fr,
                                               const float* __restrict__ pep,
                                               const float* __restrict__ pej) {
  int c = threadIdx.x;
  long row = blockIdx.x;                 // (b*75+i)*75+j
  int rem = (int)(row % 5625);
  int i = rem / 75, j = rem % 75;
  float pi = pep[(i / 25) * 128 + c] + pej[(i % 25) * 128 + c];
  float pj = pep[(j / 25) * 128 + c] + pej[(j % 25) * 128 + c];
  fr[row * 128 + c] += pi + pj;
}

// --------- attention scores: (Jq.Jk + rconv + Rq.Rk)*scale, softmax over j --
__global__ __launch_bounds__(128) void attn_scores(
    const float* __restrict__ jqkv, const float* __restrict__ rqk,
    const float* __restrict__ rconv, float* __restrict__ attn) {
  int bi = blockIdx.x;                   // (b*8+h)*75 + i
  int i  = bi % 75;
  int bh = bi / 75;
  int h  = bh & 7, b = bh >> 3;
  int j  = threadIdx.x;
  __shared__ float sc[128];
  float v = -1e30f, e = 0.f;
  if (j < 75) {
    const float* q = jqkv + (long)(b * 75 + i) * 384 + h * 16;
    const float* k = jqkv + (long)(b * 75 + j) * 384 + 128 + h * 16;
    float aJ = 0.f;
#pragma unroll
    for (int c = 0; c < 16; ++c) aJ += q[c] * k[c];
    long pr = (long)(b * 75 + i) * 75 + j;
    float aL = rconv[pr * 8 + h];
    const float* rq = rqk + pr * 256 + h * 16;
    const float* rk = rqk + pr * 256 + 128 + h * 16;
    float aQ = 0.f;
#pragma unroll
    for (int c = 0; c < 16; ++c) aQ += rq[c] * rk[c];
    v = (aJ + aL + aQ) * 0.25f;          // HS^-0.5 = 0.25
  }
  sc[j] = v; __syncthreads();
  for (int s = 64; s > 0; s >>= 1) { if (j < s) sc[j] = fmaxf(sc[j], sc[j + s]); __syncthreads(); }
  float mx = sc[0]; __syncthreads();
  e = (j < 75) ? __expf(v - mx) : 0.f;
  sc[j] = e; __syncthreads();
  for (int s = 64; s > 0; s >>= 1) { if (j < s) sc[j] += sc[j + s]; __syncthreads(); }
  float sum = sc[0];
  if (j < 75) attn[(long)bi * 75 + j] = e / sum;
}

// --------- ctx[b,i,c] = sum_j attn[b,h,i,j] * V[b,j,c] ----------------------
__global__ __launch_bounds__(128) void attn_v(const float* __restrict__ attn,
                                              const float* __restrict__ jqkv,
                                              float* __restrict__ ctx) {
  long bi = blockIdx.x;                  // b*75+i
  int b = (int)(bi / 75), i = (int)(bi % 75);
  int c = threadIdx.x;
  int h = c >> 4;
  const float* arow = attn + ((long)(b * 8 + h) * 75 + i) * 75;
  float s = 0.f;
  for (int j = 0; j < 75; ++j)
    s += arow[j] * jqkv[(long)(b * 75 + j) * 384 + 256 + c];
  ctx[bi * 128 + c] = s;
}

// ---------------------------------------------------------------------------
extern "C" void kernel_launch(void* const* d_in, const int* in_sizes, int n_in,
                              void* d_out, int out_size, void* d_ws, size_t ws_size,
                              hipStream_t stream) {
  (void)in_sizes; (void)n_in; (void)out_size; (void)ws_size;
  auto F = [&](int i) { return (const float*)d_in[i]; };

  // --- input flattening (depth-first insertion order of setup_inputs dict) ---
  // 0:x_joint 1:x_relation  je:2..7  re:8..13  pe_person:14 pe_joint:15
  // block k base bb=16+30k: jqkv(w,b)=+0,+1 rconv=+2,+3 rqk=+4,+5 proj=+6,+7
  //   na1=+8,+9 na2=+10,+11 nj=+12,+13 nr1=+14,+15 nr2=+16,+17
  //   mj(w1,b1,w2,b2)=+18..21 mr1=+22..25 mr2=+26..29
  // jd:136..141  rd:142..147
  const long CH = 18000;                 // relation-side M chunk (mult of 16)
  const long AJ = 16L * 75 * 90;         // aux_j elements
  const long AR = 16L * 75 * 75 * 30;    // aux_r elements
  float* out = (float*)d_out;
  float* tj[5]; float* tr[5];
  for (int k = 0; k < 5; ++k) {
    tj[k] = out + AJ + AR + (long)k * AJ;
    tr[k] = out + AJ + AR + 5 * AJ + (long)k * AR;
  }

  // --- workspace bump allocator ---
  char* wp = (char*)d_ws;
  auto alloc = [&](size_t bytes) -> void* {
    void* r = (void*)wp;
    wp += (bytes + 255) & ~(size_t)255;
    return r;
  };

  // --- weight conversion table ---
  int wi_idx[52], wi_K[52], wi_N[52], wKpad[52];
  _Float16* wt[52];
  int nw = 0;
  auto addw = [&](int i, int K, int N) { wi_idx[nw] = i; wi_K[nw] = K; wi_N[nw] = N; ++nw; };
  addw(2, 96, 256);  addw(4, 256, 256);  addw(6, 256, 128);   // je  -> slots 0..2
  addw(8, 18, 256);  addw(10, 256, 256); addw(12, 256, 128);  // re  -> slots 3..5
  for (int k = 0; k < 4; ++k) {                               // blocks -> 6+10k ..
    int bb = 16 + 30 * k;
    addw(bb + 0, 128, 384);  // jqkv
    addw(bb + 2, 128, 8);    // rconv
    addw(bb + 4, 128, 256);  // rqk
    addw(bb + 6, 128, 128);  // proj
    addw(bb + 18, 128, 128); // mj.w1
    addw(bb + 20, 128, 128); // mj.w2
    addw(bb + 22, 512, 512); // mr1.w1
    addw(bb + 24, 512, 128); // mr1.w2
    addw(bb + 26, 128, 128); // mr2.w1
    addw(bb + 28, 128, 128); // mr2.w2
  }
  addw(136, 128, 1024); addw(138, 1024, 512); addw(140, 512, 90);  // jd -> 46..48
  addw(142, 128, 1024); addw(144, 1024, 512); addw(146, 512, 30);  // rd -> 49..51

  for (int w = 0; w < nw; ++w) {
    int Kp = (wi_K[w] + 31) & ~31, Np = (wi_N[w] + 15) & ~15;
    wKpad[w] = Kp;
    wt[w] = (_Float16*)alloc((size_t)Kp * Np * sizeof(_Float16));
    long tot = (long)Kp * Np;
    prep_weight<<<(int)((tot + 255) / 256), 256, 0, stream>>>(
        F(wi_idx[w]), wt[w], wi_K[w], wi_N[w], Kp, Np);
  }

  // gemm: A row stride must equal wKpad[slot] (true everywhere after padding)
  auto gemm = [&](const float* A, int slot, const float* bias, const float* resid,
                  float* C, long M, int N, int act) {
    long tiles = (M >> 4) * (long)((N + 15) >> 4);
    gemm_f16_wmma<<<(int)((tiles + 7) / 8), 256, 0, stream>>>(
        A, wt[slot], bias, resid, C, (int)M, N, wKpad[slot], act);
  };

  // --- activation buffers ---
  float* fj       = (float*)alloc(1200L * 128 * 4);
  float* ln_j     = (float*)alloc(1200L * 128 * 4);
  float* jqkv_o   = (float*)alloc(1200L * 384 * 4);
  float* ctx      = (float*)alloc(1200L * 128 * 4);
  float* jh1      = (float*)alloc(1200L * 1024 * 4);
  float* jh2      = (float*)alloc(1200L * 512 * 4);
  float* fr       = (float*)alloc(90000L * 128 * 4);
  float* ln_r     = (float*)alloc(90000L * 128 * 4);
  float* rqk_o    = (float*)alloc(90000L * 256 * 4);
  float* rconv_o  = (float*)alloc(90000L * 8 * 4);
  float* attn     = (float*)alloc(16L * 8 * 75 * 75 * 4);
  float* relln    = (float*)alloc(CH * 512 * 4);
  float* rh1      = (float*)alloc(CH * 512 * 4);
  float* rhf      = (float*)alloc(90000L * 128 * 4);
  float* rc1      = (float*)alloc(CH * 1024 * 4);
  float* rc2      = (float*)alloc(CH * 512 * 4);
  float* xrel_pad = (float*)alloc(90000L * 32 * 4);

  // pad x_relation rows 18 -> 32 so the re-encoder GEMM has K=32
  {
    long tot = 90000L * 32;
    pad_rows_18_32<<<(int)((tot + 255) / 256), 256, 0, stream>>>(F(1), xrel_pad, 90000L);
  }

  auto decode_j = [&](const float* src, float* dst) {
    gemm(src, 46, F(137), nullptr, jh1, 1200, 1024, 0);
    gemm(jh1, 47, F(139), nullptr, jh2, 1200, 512, 0);
    gemm(jh2, 48, F(141), nullptr, dst, 1200, 90, 0);
  };
  auto decode_r = [&](const float* src, float* dst) {
    for (long cs = 0; cs < 90000; cs += CH) {
      gemm(src + cs * 128, 49, F(143), nullptr, rc1, CH, 1024, 0);
      gemm(rc1, 50, F(145), nullptr, rc2, CH, 512, 0);
      gemm(rc2, 51, F(147), nullptr, dst + cs * 30, CH, 30, 0);
    }
  };

  // ---------------- encoders ----------------
  gemm(F(0), 0, F(3), nullptr, jh1, 1200, 256, 0);
  gemm(jh1, 1, F(5), nullptr, jh2, 1200, 256, 0);
  gemm(jh2, 2, F(7), nullptr, fj, 1200, 128, 0);
  for (long cs = 0; cs < 90000; cs += CH) {
    gemm(xrel_pad + cs * 32, 3, F(9), nullptr, rc1, CH, 256, 0);
    gemm(rc1, 4, F(11), nullptr, rc2, CH, 256, 0);
    gemm(rc2, 5, F(13), nullptr, fr + cs * 128, CH, 128, 0);
  }

  decode_j(fj, tj[0]);
  decode_r(fr, tr[0]);

  // ---------------- transformer blocks ----------------
  for (int k = 0; k < 4; ++k) {
    int bb = 16 + 30 * k;
    int sj = 6 + 10 * k;  // slots: jqkv, rconv, rqk, proj, mjw1, mjw2, mr1w1, mr1w2, mr2w1, mr2w2

    addpe_j<<<1200, 128, 0, stream>>>(fj, F(14), F(15));
    addpe_r<<<90000, 128, 0, stream>>>(fr, F(14), F(15));

    // attention
    ln_128<<<1200, 128, 0, stream>>>(fj, F(bb + 8), F(bb + 9), ln_j);
    ln_128<<<90000, 128, 0, stream>>>(fr, F(bb + 10), F(bb + 11), ln_r);
    gemm(ln_j, sj + 0, F(bb + 1), nullptr, jqkv_o, 1200, 384, 0);
    gemm(ln_r, sj + 2, F(bb + 5), nullptr, rqk_o, 90000, 256, 0);
    gemm(ln_r, sj + 1, F(bb + 3), nullptr, rconv_o, 90000, 8, 0);
    attn_scores<<<16 * 8 * 75, 128, 0, stream>>>(jqkv_o, rqk_o, rconv_o, attn);
    attn_v<<<1200, 128, 0, stream>>>(attn, jqkv_o, ctx);
    gemm(ctx, sj + 3, F(bb + 7), fj, fj, 1200, 128, 0);

    // joint MLP
    ln_128<<<1200, 128, 0, stream>>>(fj, F(bb + 12), F(bb + 13), ln_j);
    gemm(ln_j, sj + 4, F(bb + 19), nullptr, jh1, 1200, 128, 1);
    gemm(jh1, sj + 5, F(bb + 21), fj, fj, 1200, 128, 0);

    // relation MLP 1 (4C concat -> LN -> MLP -> residual), M-chunked
    for (long cs = 0; cs < 90000; cs += CH) {
      build_relln<<<(int)CH, 256, 0, stream>>>(fr, fj, F(bb + 14), F(bb + 15), relln, cs);
      gemm(relln, sj + 6, F(bb + 23), nullptr, rh1, CH, 512, 1);
      gemm(rh1, sj + 7, F(bb + 25), fr + cs * 128, fr + cs * 128, CH, 128, 0);
    }

    // relation MLP 2
    ln_128<<<90000, 128, 0, stream>>>(fr, F(bb + 16), F(bb + 17), ln_r);
    gemm(ln_r, sj + 8, F(bb + 27), nullptr, rhf, 90000, 128, 1);
    gemm(rhf, sj + 9, F(bb + 29), fr, fr, 90000, 128, 0);

    decode_j(fj, tj[k + 1]);
    decode_r(fr, tr[k + 1]);
  }

  // pred == last aux entries
  hipMemcpyAsync(out, tj[4], AJ * sizeof(float), hipMemcpyDeviceToDevice, stream);
  hipMemcpyAsync(out + AJ, tr[4], AR * sizeof(float), hipMemcpyDeviceToDevice, stream);
}
